// PromptDetectionLoss_42640435315170
// MI455X (gfx1250) — matrix operand order repaired and checked by reading, not compiled, not run on hardware
//
#include <hip/hip_runtime.h>
#include <hip/hip_bf16.h>
#include <math.h>

typedef __attribute__((ext_vector_type(16))) _Float16 v16h;
typedef __attribute__((ext_vector_type(8)))  float    v8f;

#define TOPK_ 8
#define GMAX 256
#define SORT_CAP 4096
#define USEDP_WORDS 2048   // bitset for up to 65536 anchors

// ---------------------------------------------------------------- helpers
__device__ __forceinline__ float sigm(float x) { return 1.0f / (1.0f + expf(-x)); }

// sorted-descending top-8 insertion (stable-ish: strict '>' keeps earlier idx first)
__device__ __forceinline__ void top8_insert(float* v, int* ix, float nv, int ni) {
  if (!(nv > v[7])) return;
  int j = 7;
  while (j > 0 && nv > v[j - 1]) { v[j] = v[j - 1]; ix[j] = ix[j - 1]; --j; }
  v[j] = nv; ix[j] = ni;
}

// merge two sorted-descending 8-lists (a,b) -> a
__device__ __forceinline__ void merge8(float* av, int* ai, const float* bv, const int* bi) {
  float ov[8]; int oi[8];
  int x = 0, y = 0;
#pragma unroll
  for (int k = 0; k < 8; ++k) {
    bool ta = (av[x] > bv[y]) || ((av[x] == bv[y]) && (ai[x] <= bi[y]));
    if (ta) { ov[k] = av[x]; oi[k] = ai[x]; ++x; }
    else    { ov[k] = bv[y]; oi[k] = bi[y]; ++y; }
  }
#pragma unroll
  for (int k = 0; k < 8; ++k) { av[k] = ov[k]; ai[k] = oi[k]; }
}

// block-wide (256 thr) top-8 reduction; result lands in shv[0..7]/shi[0..7]
__device__ void block_reduce8(float* shv, int* shi, float* v, int* ix, int tid) {
#pragma unroll
  for (int k = 0; k < 8; ++k) { shv[tid * 8 + k] = v[k]; shi[tid * 8 + k] = ix[k]; }
  __syncthreads();
  for (int s = 128; s >= 1; s >>= 1) {
    if (tid < s) {
      float a[8], b[8]; int ai[8], bi[8];
#pragma unroll
      for (int k = 0; k < 8; ++k) {
        a[k] = shv[tid * 8 + k];       ai[k] = shi[tid * 8 + k];
        b[k] = shv[(tid + s) * 8 + k]; bi[k] = shi[(tid + s) * 8 + k];
      }
      merge8(a, ai, b, bi);
#pragma unroll
      for (int k = 0; k < 8; ++k) { shv[tid * 8 + k] = a[k]; shi[tid * 8 + k] = ai[k]; }
    }
    __syncthreads();
  }
}

// ------------------------------------------------- K1: per-GT metric + top-8
__global__ __launch_bounds__(256)
void topk_kernel(const float* __restrict__ scores, const float* __restrict__ priors,
                 const float* __restrict__ boxes, const float* __restrict__ objn,
                 const float* __restrict__ tgtn, const float* __restrict__ ap,
                 const float* __restrict__ gboxes, const int* __restrict__ glab,
                 const unsigned char* __restrict__ vmask,
                 float* __restrict__ pair_score, int* __restrict__ pair_p,
                 int N, int C) {
  __shared__ float shv[256 * 8];
  __shared__ int   shi[256 * 8];
  __shared__ float finv[8];
  __shared__ int   fini[8];
  __shared__ int   sAny;

  const int g = blockIdx.x, tid = threadIdx.x;
  if (tid == 0) sAny = 0;

  const float4 gb = reinterpret_cast<const float4*>(gboxes)[g];
  const int gl = glab[g];
  const int lab = gl < 0 ? 0 : (gl > C - 1 ? C - 1 : gl);
  const bool valid = (gl >= 0) && (gl < C) && (vmask[lab] != 0);
  const float gcx = (gb.x + gb.z) * 0.5f, gcy = (gb.y + gb.w) * 0.5f;
  const float rhx = 1.0f / fmaxf((gb.z - gb.x) * 0.5f, 1.0f);
  const float rhy = 1.0f / fmaxf((gb.w - gb.y) * 0.5f, 1.0f);
  const float garea = fmaxf(gb.z - gb.x, 0.f) * fmaxf(gb.w - gb.y, 0.f);

  const float NINF = -__builtin_inff();
  float sv[8], lv[8]; int si_[8], li_[8];
#pragma unroll
  for (int k = 0; k < 8; ++k) { sv[k] = NINF; lv[k] = NINF; si_[k] = 0; li_[k] = 0; }

  __syncthreads();  // sAny init visible

  int myAny = 0;
  if (valid) {
    for (int n = tid; n < N; n += 256) {
      const float axp = ap[2 * n], ayp = ap[2 * n + 1];
      const bool inside = (axp >= gb.x) & (axp <= gb.z) & (ayp >= gb.y) & (ayp <= gb.w);
      if (!inside) continue;
      const float dx = fabsf(axp - gcx) * rhx;
      const float dy = fabsf(ayp - gcy) * rhy;
      const bool strict = fmaxf(dx, dy) <= 0.75f;
      myAny |= (int)strict;
      const float4 pb = reinterpret_cast<const float4*>(boxes)[n];
      const float iw = fmaxf(fminf(pb.z, gb.z) - fmaxf(pb.x, gb.x), 0.f);
      const float ih = fmaxf(fminf(pb.w, gb.w) - fmaxf(pb.y, gb.y), 0.f);
      const float inter = iw * ih;
      const float pa = fmaxf(pb.z - pb.x, 0.f) * fmaxf(pb.w - pb.y, 0.f);
      const float iou = inter / (pa + garea - inter + 1e-7f);
      const float o  = sigm(objn[n]);
      const float tg = sigm(tgtn[n]);
      const float cl = sigm(scores[(size_t)n * C + lab]);
      const float pr = fmaxf(priors[(size_t)n * C + lab], 1e-4f);
      const float q  = sqrtf(sqrtf(fmaxf(o * tg * cl * pr, 0.f)));
      const float cp = expf(-0.5f * (dx * dx + dy * dy));
      const float metric = q * (iou * iou) * (cp * cp);
      top8_insert(lv, li_, metric, n);
      if (strict) top8_insert(sv, si_, metric, n);
    }
  }
  if (myAny) atomicOr(&sAny, 1);

  block_reduce8(shv, shi, sv, si_, tid);          // strict list
  if (tid == 0) {
#pragma unroll
    for (int k = 0; k < 8; ++k) { finv[k] = shv[k]; fini[k] = shi[k]; }
  }
  __syncthreads();
  block_reduce8(shv, shi, lv, li_, tid);          // loose (inside) list
  if (tid == 0) {
    const bool us = (sAny != 0);
#pragma unroll
    for (int k = 0; k < 8; ++k) {
      pair_score[g * TOPK_ + k] = us ? finv[k] : shv[k];
      pair_p[g * TOPK_ + k]     = us ? fini[k] : shi[k];
    }
  }
}

// -------------------------- K2: bitonic sort + sequential greedy matching
__global__ __launch_bounds__(1024)
void match_kernel(const float* __restrict__ pair_score, const int* __restrict__ pair_p,
                  const int* __restrict__ glab,
                  int* __restrict__ matched_gt, int* __restrict__ matched_lab,
                  int* __restrict__ used_g_out,
                  int N, int C, int G, int SORTN) {
  __shared__ float skey[SORT_CAP];
  __shared__ int   sidx[SORT_CAP];
  __shared__ unsigned short sp[SORT_CAP];
  __shared__ unsigned short sg[SORT_CAP];
  __shared__ unsigned int usedp[USEDP_WORDS];
  __shared__ int usedg[GMAX];

  const int tid = threadIdx.x, nt = blockDim.x;
  const int GP = G * TOPK_;
  const float INF = __builtin_inff();

  for (int i = tid; i < SORTN; i += nt) {
    float key = INF;
    if (i < GP) {
      float s = pair_score[i];
      if (s > -INF && s < INF) key = -s;   // finite -> ascending key = -score
    }
    skey[i] = key; sidx[i] = i;
  }
  for (int i = tid; i < USEDP_WORDS; i += nt) usedp[i] = 0u;
  for (int i = tid; i < G; i += nt) usedg[i] = 0;
  for (int i = tid; i < N; i += nt) { matched_gt[i] = -1; matched_lab[i] = -1; }
  __syncthreads();

  for (int k = 2; k <= SORTN; k <<= 1) {
    for (int j = k >> 1; j > 0; j >>= 1) {
      for (int i = tid; i < SORTN; i += nt) {
        const int pr = i ^ j;
        if (pr > i) {
          const float k1 = skey[i], k2 = skey[pr];
          const int   i1 = sidx[i], i2 = sidx[pr];
          const bool lt = (k1 < k2) || ((k1 == k2) && (i1 < i2));
          const bool up = ((i & k) == 0);
          if (up != lt) { skey[i] = k2; skey[pr] = k1; sidx[i] = i2; sidx[pr] = i1; }
        }
      }
      __syncthreads();
    }
  }

  for (int i = tid; i < SORTN; i += nt) {      // stage p,g per rank into LDS
    const int orig = sidx[i];
    sp[i] = (unsigned short)((orig < GP) ? pair_p[orig] : 0);
    sg[i] = (unsigned short)(orig >> 3);
  }
  __syncthreads();

  if (tid == 0) {                               // greedy scan (matches lax.scan)
    for (int i = 0; i < SORTN; ++i) {
      if (skey[i] == INF) break;                // invalid pairs are last
      const int p = sp[i], g = sg[i];
      const unsigned int w = usedp[p >> 5], bit = 1u << (p & 31);
      if (!(w & bit) && !usedg[g]) {
        usedp[p >> 5] = w | bit;
        usedg[g] = 1;
        matched_gt[p] = g;
        const int gl = glab[g];
        matched_lab[p] = gl < 0 ? 0 : (gl > C - 1 ? C - 1 : gl);
      }
    }
  }
  __syncthreads();
  for (int i = tid; i < G; i += nt) used_g_out[i] = usedg[i];
}

// ------ K3: per-anchor finalize; duplicate_mask via WMMA (N x G) * used_g
__global__ __launch_bounds__(256)
void finalize_kernel(const float* __restrict__ ap, const float* __restrict__ boxes,
                     const float* __restrict__ gboxes,
                     const int* __restrict__ matched_gt, const int* __restrict__ matched_lab,
                     const int* __restrict__ used_g,
                     float* __restrict__ out_boxes, float* __restrict__ out_fg,
                     float* __restrict__ out_mgt, float* __restrict__ out_mlab,
                     float* __restrict__ out_dup, float* __restrict__ ts_val,
                     int N, int G) {
  __shared__ float4  s_box[GMAX];
  __shared__ float2  s_ctr[GMAX];
  __shared__ float2  s_rih[GMAX];     // inverse half-extents (no div in hot loop)
  __shared__ _Float16 s_u[GMAX];
  __shared__ float   s_dup[8][16];

  const int tid = threadIdx.x;
  for (int g = tid; g < GMAX; g += 256) {
    if (g < G) {
      const float4 b = reinterpret_cast<const float4*>(gboxes)[g];
      s_box[g]  = b;
      s_ctr[g]  = make_float2((b.x + b.z) * 0.5f, (b.y + b.w) * 0.5f);
      s_rih[g]  = make_float2(1.0f / fmaxf((b.z - b.x) * 0.5f, 1.f),
                              1.0f / fmaxf((b.w - b.y) * 0.5f, 1.f));
      s_u[g]    = used_g[g] ? (_Float16)1.0f : (_Float16)0.0f;
    } else {
      s_box[g]  = make_float4(0.f, 0.f, -1.f, -1.f);   // 'inside' always false
      s_ctr[g]  = make_float2(0.f, 0.f);
      s_rih[g]  = make_float2(1.f, 1.f);
      s_u[g]    = (_Float16)0.0f;
    }
  }
  __syncthreads();

  const int wave = tid >> 5, lane = tid & 31;
  const int nbase = (blockIdx.x * 8 + wave) * 16;
  const int row = lane & 15;
  int n = nbase + row; if (n > N - 1) n = N - 1;     // keep EXEC all-1s for WMMA
  const float axp = ap[2 * n], ayp = ap[2 * n + 1];
  const int kOffA = (lane < 16) ? 0 : 8;             // A 16x32 f16 lane layout
  const int kOffB = (lane < 16) ? 0 : 16;            // unused: B is read with A's k order (gA)

  v8f acc = {0.f, 0.f, 0.f, 0.f, 0.f, 0.f, 0.f, 0.f};
#pragma unroll
  for (int c = 0; c < 8; ++c) {                      // 8 chunks of 32 GTs = 256
    const int g0 = c * 32;
    v16h A, B;
#pragma unroll
    for (int v = 0; v < 8; ++v) {
#pragma unroll
      for (int h = 0; h < 2; ++h) {
        const int e = 2 * v + h;
        const int kA = ((v < 4) ? (2 * v + h) : (16 + 2 * (v - 4) + h)) + kOffA;
        const int gA = g0 + kA;
        const float4 b = s_box[gA];
        // branch-free 0/1 mask: bitwise AND of compares -> v_cmp/v_cndmask
        const int ins = (int)(axp >= b.x) & (int)(axp <= b.z) &
                        (int)(ayp >= b.y) & (int)(ayp <= b.w);
        const float dx = fabsf(axp - s_ctr[gA].x) * s_rih[gA].x;
        const float dy = fabsf(ayp - s_ctr[gA].y) * s_rih[gA].y;
        const int m = ins & (int)(fmaxf(dx, dy) <= 1.25f);
        A[e] = (_Float16)(float)m;
        B[e] = s_u[gA];                              // used_g replicated columns
      }
    }
    acc = __builtin_amdgcn_wmma_f32_16x16x32_f16(false, A, false, B,
                                                 (short)0, acc, false, false);
  }
  // D column 0: lane0 holds rows 0..7, lane16 holds rows 8..15
  if (lane == 0)  { for (int j = 0; j < 8; ++j) s_dup[wave][j]     = acc[j]; }
  if (lane == 16) { for (int j = 0; j < 8; ++j) s_dup[wave][8 + j] = acc[j]; }
  __syncthreads();

  if (tid < 128) {
    const int n2 = blockIdx.x * 128 + tid;
    if (n2 < N) {
      const int mg = matched_gt[n2];
      const bool fg = mg >= 0;
      float ts = 0.f;
      float4 tb = make_float4(0.f, 0.f, 0.f, 0.f);
      if (fg) {
        const float4 pb = reinterpret_cast<const float4*>(boxes)[n2];
        const float4 gbb = s_box[mg];
        const float iw = fmaxf(fminf(pb.z, gbb.z) - fmaxf(pb.x, gbb.x), 0.f);
        const float ih = fmaxf(fminf(pb.w, gbb.w) - fmaxf(pb.y, gbb.y), 0.f);
        const float inter = iw * ih;
        const float pa = fmaxf(pb.z - pb.x, 0.f) * fmaxf(pb.w - pb.y, 0.f);
        const float ga = fmaxf(gbb.z - gbb.x, 0.f) * fmaxf(gbb.w - gbb.y, 0.f);
        const float iou = inter / (pa + ga - inter + 1e-7f);
        ts = fmaxf(iou, 0.1f);
        tb = gbb;
      }
      const bool dup = (!fg) && (s_dup[tid >> 4][tid & 15] > 0.5f);
      reinterpret_cast<float4*>(out_boxes)[n2] = tb;
      out_fg[n2]   = fg ? 1.f : 0.f;
      out_mgt[n2]  = (float)mg;
      out_mlab[n2] = (float)matched_lab[n2];
      out_dup[n2]  = dup ? 1.f : 0.f;
      ts_val[n2]   = ts;
    }
  }
}

// --------------------- K4: target_scores fill (store-bandwidth bound, 162MB)
// Row-per-block; aligned body uses float4 (global_store_b128).
__global__ __launch_bounds__(256)
void scores_kernel(const int* __restrict__ matched_lab, const float* __restrict__ ts_val,
                   float* __restrict__ out, int C) {
  const int n = blockIdx.x;
  const int lab = matched_lab[n];          // -1 for background -> all zeros
  const float ts = ts_val[n];
  float* row = out + (size_t)n * C;

  // scalar head to reach 16B alignment
  const int head = (int)(((16u - ((unsigned)(uintptr_t)row & 15u)) & 15u) >> 2);
  for (int c = threadIdx.x; c < head && c < C; c += blockDim.x)
    row[c] = (c == lab) ? ts : 0.0f;

  const int nvec = (C > head) ? ((C - head) >> 2) : 0;
  for (int i = threadIdx.x; i < nvec; i += blockDim.x) {
    const int c0 = head + 4 * i;
    float4 vv = make_float4(0.f, 0.f, 0.f, 0.f);
    if (lab >= c0 && lab < c0 + 4) {
      float* pv = reinterpret_cast<float*>(&vv);
      pv[lab - c0] = ts;
    }
    *reinterpret_cast<float4*>(row + c0) = vv;
  }

  for (int c = head + 4 * nvec + threadIdx.x; c < C; c += blockDim.x)
    row[c] = (c == lab) ? ts : 0.0f;
}

// ----------------------------------------------------------------- launcher
extern "C" void kernel_launch(void* const* d_in, const int* in_sizes, int n_in,
                              void* d_out, int out_size, void* d_ws, size_t ws_size,
                              hipStream_t stream) {
  const float* scores = (const float*)d_in[0];
  const float* priors = (const float*)d_in[1];
  const float* boxes  = (const float*)d_in[2];
  const float* objn   = (const float*)d_in[3];
  const float* tgtn   = (const float*)d_in[4];
  const float* ap     = (const float*)d_in[5];
  const float* gboxes = (const float*)d_in[6];
  const int*   glab   = (const int*)d_in[7];
  const unsigned char* vmask = (const unsigned char*)d_in[8];

  const int N  = in_sizes[2] / 4;          // pred_boxes (N,4)
  const int C  = in_sizes[0] / N;          // pred_scores (N,C)
  const int G  = in_sizes[7];              // gt_labels (G,)
  const int GP = G * TOPK_;

  float* out        = (float*)d_out;
  float* out_scores = out;                              // N*C
  float* out_boxes  = out + (size_t)N * C;              // N*4
  float* out_fg     = out_boxes + (size_t)N * 4;        // N
  float* out_mgt    = out_fg + N;                       // N
  float* out_mlab   = out_mgt + N;                      // N
  float* out_dup    = out_mlab + N;                     // N

  float* ws_score  = (float*)d_ws;                      // GP floats
  int*   ws_pair_p = (int*)d_ws + GP;                   // GP ints
  int*   ws_mgt    = (int*)d_ws + 2 * GP;               // N
  int*   ws_mlab   = ws_mgt + N;                        // N
  int*   ws_usedg  = ws_mlab + N;                       // G
  float* ws_ts     = (float*)(void*)(ws_usedg + G);     // N

  topk_kernel<<<G, 256, 0, stream>>>(scores, priors, boxes, objn, tgtn, ap,
                                     gboxes, glab, vmask, ws_score, ws_pair_p, N, C);

  int SORTN = 2;
  while (SORTN < GP) SORTN <<= 1;
  if (SORTN > SORT_CAP) SORTN = SORT_CAP;
  match_kernel<<<1, 1024, 0, stream>>>(ws_score, ws_pair_p, glab,
                                       ws_mgt, ws_mlab, ws_usedg, N, C, G, SORTN);

  const int blocks3 = (N + 127) / 128;
  finalize_kernel<<<blocks3, 256, 0, stream>>>(ap, boxes, gboxes, ws_mgt, ws_mlab,
                                               ws_usedg, out_boxes, out_fg, out_mgt,
                                               out_mlab, out_dup, ws_ts, N, G);

  scores_kernel<<<N, 256, 0, stream>>>(ws_mlab, ws_ts, out_scores, C);
}
